// MeshGCN_65343632441451
// MI455X (gfx1250) — matrix operand order repaired and checked by reading, not compiled, not run on hardware
//
#include <hip/hip_runtime.h>
#include <hip/hip_bf16.h>
#include <stdint.h>

typedef __attribute__((ext_vector_type(16))) _Float16 v16h;
typedef __attribute__((ext_vector_type(8)))  _Float16 v8h;
typedef __attribute__((ext_vector_type(8)))  float    v8f;

#define HCH   24   // hidden channels
#define KPAD  32   // padded K for one 16x16x32 WMMA
#define OUTC  3

// ---------------------------------------------------------------------------
// 1) Build node features x[N,24] from featr3 / stmdist, store f16 padded [N,32]
// ---------------------------------------------------------------------------
__global__ void build_x_kernel(const float* __restrict__ featr3,
                               const float* __restrict__ stmdist,
                               _Float16* __restrict__ xh, int N) {
  int n = blockIdx.x * blockDim.x + threadIdx.x;
  if (n >= N) return;
  const float* f = featr3 + (size_t)n * 27;
  const float* s = stmdist + (size_t)n * 3;
  _Float16* o = xh + (size_t)n * KPAD;
  // upper-tri (r,c) pairs (0,0)(0,1)(0,2)(1,1)(1,2)(2,2) -> flat 0,1,2,4,5,8
  const int ut[6] = {0, 1, 2, 4, 5, 8};
#pragma unroll
  for (int j = 0; j < 6; ++j) o[j]     = (_Float16)f[ut[j]];       // plane 0
#pragma unroll
  for (int j = 0; j < 6; ++j) o[6 + j] = (_Float16)f[9 + ut[j]];   // plane 1
#pragma unroll
  for (int j = 0; j < 9; ++j) o[12 + j] = (_Float16)f[18 + j];     // plane 2
#pragma unroll
  for (int j = 0; j < 3; ++j) o[21 + j] = (_Float16)s[j];
#pragma unroll
  for (int j = HCH; j < KPAD; ++j) o[j] = (_Float16)0.0f;
}

// ---------------------------------------------------------------------------
// 2) GCN normalization: deg (self-loop => init 1), atomic accumulate, rsqrt
// ---------------------------------------------------------------------------
__global__ void deg_init_kernel(float* __restrict__ deg, int N) {
  int i = blockIdx.x * blockDim.x + threadIdx.x;
  if (i < N) deg[i] = 1.0f;   // self loop
}

__global__ void deg_accum_kernel(const long long* __restrict__ ei,
                                 float* __restrict__ deg, int E) {
  int e = blockIdx.x * blockDim.x + threadIdx.x;
  if (e >= E) return;
  long long d = ei[(size_t)E + e];  // dst row
  __hip_atomic_fetch_add(&deg[d], 1.0f, __ATOMIC_RELAXED, __HIP_MEMORY_SCOPE_AGENT);
}

__global__ void rsqrt_kernel(float* __restrict__ deg, int N) {
  int i = blockIdx.x * blockDim.x + threadIdx.x;
  if (i < N) deg[i] = rsqrtf(deg[i]);  // deg >= 1 always (self loop)
}

// ---------------------------------------------------------------------------
// 3) Pack weights into per-lane WMMA B-fragment order: bpk[tile][lane][16]
//    element j of lane l <-> B[k = 8*(l/16) + j (j<8) | 16 + 8*(l/16) + j-8 (j>=8)][n = l%16] of the tile.
//    tiles 0,1: Wh[4] cols 0..15 / 16..23 (zero padded); tile 2: W_out.
// ---------------------------------------------------------------------------
__global__ void pack_w_kernel(const float* __restrict__ Wh,
                              const float* __restrict__ Wout,
                              _Float16* __restrict__ bpk) {
  int t = blockIdx.x;        // tile 0..2
  int r = threadIdx.x;       // 0..511
  int l = r >> 4;
  int j = r & 15;
  int n = l & 15;
  int k = (j < 8) ? ((l >> 4) << 3) + j : 16 + ((l >> 4) << 3) + (j - 8);
  float v = 0.0f;
  if (t < 2) {
    int col = t * 16 + n;
    if (k < HCH && col < HCH)
      v = Wh[(size_t)4 * HCH * HCH + (size_t)k * HCH + col];  // layer 4 only (0-3 are dead)
  } else {
    if (k < HCH && n < OUTC)
      v = Wout[(size_t)k * OUTC + n];
  }
  bpk[(size_t)t * 512 + r] = (_Float16)v;
}

// ---------------------------------------------------------------------------
// 4) Dense transform h = x @ W via V_WMMA_F32_16X16X32_F16, wave32.
//    One wave -> 16 rows; NT tiles of 16 cols; store COLS valid columns.
//    Epilogue: fast path for full stripes (guards hoisted out of row loop).
// ---------------------------------------------------------------------------
template <int NT, int COLS>
__global__ void gemm_wmma_kernel(const _Float16* __restrict__ xh,
                                 const _Float16* __restrict__ bpk,
                                 float* __restrict__ out, int N) {
  int lane  = threadIdx.x & 31;
  int wave  = threadIdx.x >> 5;
  int m0    = (blockIdx.x * (blockDim.x >> 5) + wave) * 16;
  if (m0 >= N) return;                 // wave-uniform; EXEC all-ones for WMMA
  int hi = lane >> 4;                  // lane half
  int k0 = hi * 8;

  // A fragment: two contiguous 16B chunks per lane (ISA 16-bit A 16x32 layout)
  v16h a;
  {
    int row = m0 + (lane & 15);
    if (row >= N) row = N - 1;         // clamp loads; stores are guarded
    const _Float16* rp = xh + (size_t)row * KPAD;
    v8h lo = *(const v8h*)(rp + k0);
    v8h up = *(const v8h*)(rp + 16 + k0);
#pragma unroll
    for (int q = 0; q < 8; ++q) { a[q] = lo[q]; a[8 + q] = up[q]; }
  }

  v8f acc[NT];
#pragma unroll
  for (int t = 0; t < NT; ++t) {
    v16h b = *(const v16h*)(bpk + (size_t)t * 512 + (size_t)lane * 16);
    v8f c = {};
    acc[t] = __builtin_amdgcn_wmma_f32_16x16x32_f16(
        false, a, false, b, (short)0, c, false, false);
  }

  int n0 = lane & 15;
  int r0 = m0 + hi * 8;                // first row this lane writes
  if (m0 + 16 <= N) {
    // Full stripe: hoist the (lane-uniform) column guard out of the row loop.
#pragma unroll
    for (int t = 0; t < NT; ++t) {
      int col = t * 16 + n0;
      if (col < COLS) {
        float* p = out + (size_t)r0 * COLS + col;
#pragma unroll
        for (int r = 0; r < 8; ++r) p[(size_t)r * COLS] = acc[t][r];
      }
    }
  } else {
    // Ragged tail stripe (only when N % 16 != 0)
#pragma unroll
    for (int r = 0; r < 8; ++r) {
      int row = r0 + r;
      if (row >= N) continue;
#pragma unroll
      for (int t = 0; t < NT; ++t) {
        int col = t * 16 + n0;
        if (col < COLS) out[(size_t)row * COLS + col] = acc[t][r];
      }
    }
  }
}

// ---------------------------------------------------------------------------
// 5) agg[n][c] = bias[c] + h[n][c] * dis[n]^2   (bias + self-loop term)
// ---------------------------------------------------------------------------
template <int C>
__global__ void init_agg_kernel(const float* __restrict__ h,
                                const float* __restrict__ dis,
                                const float* __restrict__ bias,
                                float* __restrict__ agg, int N) {
  int idx = blockIdx.x * blockDim.x + threadIdx.x;
  if (idx >= N * C) return;
  int n = idx / C, c = idx - n * C;
  float di = dis[n];
  agg[idx] = bias[c] + h[idx] * di * di;
}

// ---------------------------------------------------------------------------
// 6) Edge scatter: agg[dst] += h[src] * dis[src]*dis[dst]  (fp32 atomics).
//    Prefetch the source row so its cacheline fetch overlaps the dis gathers.
// ---------------------------------------------------------------------------
template <int C>
__global__ void edge_agg_kernel(const long long* __restrict__ ei,
                                const float* __restrict__ h,
                                const float* __restrict__ dis,
                                float* __restrict__ agg, int E) {
  int e = blockIdx.x * blockDim.x + threadIdx.x;
  if (e >= E) return;
  long long s = ei[e];
  long long d = ei[(size_t)E + e];
  const float* hp = h + (size_t)s * C;
  __builtin_prefetch(hp, 0, 3);        // global_prefetch_b8: overlap with dis gathers
  float nm = dis[s] * dis[d];
  float* op = agg + (size_t)d * C;
#pragma unroll
  for (int c = 0; c < C; ++c) {
    __hip_atomic_fetch_add(&op[c], hp[c] * nm,
                           __ATOMIC_RELAXED, __HIP_MEMORY_SCOPE_AGENT);
  }
}

// ---------------------------------------------------------------------------
// 7) relu + convert to f16 padded [N,32] (input for output-layer GEMM)
// ---------------------------------------------------------------------------
__global__ void relu_cvt_kernel(const float* __restrict__ agg,
                                _Float16* __restrict__ xh, int N) {
  int idx = blockIdx.x * blockDim.x + threadIdx.x;
  if (idx >= N * KPAD) return;
  int n = idx / KPAD, c = idx - n * KPAD;
  float v = 0.0f;
  if (c < HCH) {
    v = agg[(size_t)n * HCH + c];
    v = v > 0.0f ? v : 0.0f;
  }
  xh[idx] = (_Float16)v;
}

// ---------------------------------------------------------------------------
extern "C" void kernel_launch(void* const* d_in, const int* in_sizes, int n_in,
                              void* d_out, int out_size, void* d_ws, size_t ws_size,
                              hipStream_t stream) {
  const float*     featr3  = (const float*)d_in[0];
  const float*     stmdist = (const float*)d_in[1];
  const long long* ei      = (const long long*)d_in[2];  // int64 [2,E]
  const float*     Wh      = (const float*)d_in[3];      // [5,24,24]
  const float*     bh      = (const float*)d_in[4];      // [5,24]
  const float*     Wout    = (const float*)d_in[5];      // [24,3]
  const float*     bout    = (const float*)d_in[6];      // [3]
  float*           outp    = (float*)d_out;              // [N,3]

  const int N = in_sizes[1] / 3;
  const int E = in_sizes[2] / 2;

  // Workspace layout (256B aligned slices)
  char* ws = (char*)d_ws;
  size_t off = 0;
  auto take = [&](size_t bytes) -> char* {
    char* p = ws + off;
    off = (off + bytes + 255) & ~(size_t)255;
    return p;
  };
  float*    dis  = (float*)take((size_t)N * sizeof(float));
  _Float16* xh   = (_Float16*)take((size_t)N * KPAD * sizeof(_Float16));
  float*    hbuf = (float*)take((size_t)N * HCH * sizeof(float));   // h, later reused as g[N,3]
  float*    agg  = (float*)take((size_t)N * HCH * sizeof(float));
  _Float16* bpk  = (_Float16*)take((size_t)3 * 512 * sizeof(_Float16));
  float*    g    = hbuf;  // output-layer pre-aggregation, aliases h (h dead by then)

  const int BT = 256;
  auto blks = [&](long long work) { return (int)((work + BT - 1) / BT); };

  // Feature build + degree norm (independent chains, same stream)
  build_x_kernel<<<blks(N), BT, 0, stream>>>(featr3, stmdist, xh, N);
  deg_init_kernel<<<blks(N), BT, 0, stream>>>(dis, N);
  deg_accum_kernel<<<blks(E), BT, 0, stream>>>(ei, dis, E);
  rsqrt_kernel<<<blks(N), BT, 0, stream>>>(dis, N);
  pack_w_kernel<<<3, 512, 0, stream>>>(Wh, Wout, bpk);

  // Hidden layer (only layer 4 is live): h = x @ Wh[4]  (2 col-tiles)
  {
    int stripes = (N + 15) / 16;            // one wave per 16-row stripe
    int grid = (stripes + (BT / 32) - 1) / (BT / 32);
    gemm_wmma_kernel<2, HCH><<<grid, BT, 0, stream>>>(xh, bpk, hbuf, N);
  }
  init_agg_kernel<HCH><<<blks((long long)N * HCH), BT, 0, stream>>>(
      hbuf, dis, bh + 4 * HCH, agg, N);
  edge_agg_kernel<HCH><<<blks(E), BT, 0, stream>>>(ei, hbuf, dis, agg, E);

  // relu -> f16, output layer GEMM (1 col-tile, 3 valid cols)
  relu_cvt_kernel<<<blks((long long)N * KPAD), BT, 0, stream>>>(agg, xh, N);
  {
    int stripes = (N + 15) / 16;
    int grid = (stripes + (BT / 32) - 1) / (BT / 32);
    gemm_wmma_kernel<1, OUTC><<<grid, BT, 0, stream>>>(xh, bpk + 2 * 512, g, N);
  }
  init_agg_kernel<OUTC><<<blks((long long)N * OUTC), BT, 0, stream>>>(
      g, dis, bout, outp, N);
  edge_agg_kernel<OUTC><<<blks(E), BT, 0, stream>>>(ei, g, dis, outp, E);
}
